// GridSumAttention_59846074302961
// MI455X (gfx1250) — matrix operand order repaired and checked by reading, not compiled, not run on hardware
//
#include <hip/hip_runtime.h>
#include <hip/hip_bf16.h>

typedef _Float16 f16;
typedef __attribute__((ext_vector_type(16))) _Float16 v16h;
typedef __attribute__((ext_vector_type(8)))  _Float16 h8;
typedef __attribute__((ext_vector_type(8)))  float    v8f;

#define NHG 36   // gathered neighborhood size

__device__ inline void frag_put2(v16h& f, int v, unsigned u) {
    union { unsigned u; _Float16 h[2]; } cv; cv.u = u;
    f[2 * v]     = cv.h[0];
    f[2 * v + 1] = cv.h[1];
}

// 16-byte async global->LDS copy (CDNA5 GLOBAL_LOAD_ASYNC_TO_LDS_B128,
// tracked by ASYNCcnt; per-lane LDS destination address).
__device__ inline void async_ld16(const f16* gsrc, _Float16* lds_dst) {
    unsigned loff = (unsigned)(size_t)(void*)lds_dst;
    asm volatile("global_load_async_to_lds_b128 %0, %1, off"
                 :: "v"(loff), "v"(gsrc)
                 : "memory");
}

__device__ inline void async_wait0() {
    asm volatile("s_wait_asynccnt 0x0" ::: "memory");
}

// ---------------------------------------------------------------------------
// f32 -> f16 weight conversion
// ---------------------------------------------------------------------------
__global__ void f32_to_f16_kernel(const float* __restrict__ src,
                                  f16* __restrict__ dst, int n) {
    int i = blockIdx.x * 256 + threadIdx.x;
    if (i < n) dst[i] = (_Float16)src[i];
}

// ---------------------------------------------------------------------------
// Dual LayerNorm: one wave (32 lanes) per token, C = 384 (12 elems / lane).
// ---------------------------------------------------------------------------
template <bool DUAL>
__global__ __launch_bounds__(256)
void ln_dual_kernel(const float* __restrict__ x,
                    const float* __restrict__ s1, const float* __restrict__ b1,
                    const float* __restrict__ s2, const float* __restrict__ b2,
                    f16* __restrict__ o1, f16* __restrict__ o2, int M) {
    const int C = 384;
    int wid  = threadIdx.x >> 5;
    int lane = threadIdx.x & 31;
    int token = blockIdx.x * 8 + wid;
    if (token >= M) return;
    const float* xr = x + (size_t)token * C;

    float vals[12];
    float s = 0.f, sq = 0.f;
#pragma unroll
    for (int i = 0; i < 12; ++i) {
        float v = xr[lane + i * 32];
        vals[i] = v; s += v; sq += v * v;
    }
#pragma unroll
    for (int off = 16; off > 0; off >>= 1) {
        s  += __shfl_xor(s,  off, 32);
        sq += __shfl_xor(sq, off, 32);
    }
    float mean = s * (1.f / 384.f);
    float var  = sq * (1.f / 384.f) - mean * mean;
    float rstd = rsqrtf(var + 1e-5f);

#pragma unroll
    for (int i = 0; i < 12; ++i) {
        int c = lane + i * 32;
        float xh = (vals[i] - mean) * rstd;
        o1[(size_t)token * C + c] = (_Float16)(xh * s1[c] + b1[c]);
        if (DUAL) o2[(size_t)token * C + c] = (_Float16)(xh * s2[c] + b2[c]);
    }
}

// ---------------------------------------------------------------------------
// Generic f16 WMMA GEMM: C[M,N] = A[M,K] @ B[K,N] (+bias) with fused epilogue.
//  MODE 0: out16 = acc (+bias)
//  MODE 1: out16 = gelu(acc + bias)
//  MODE 2: out32 = res32 + acc
//  MODE 3: out32 = res32 + gamma[col] * (acc + bias)
// Block = 256 threads (8 waves), tile 64x64, K-step 32.
// ---------------------------------------------------------------------------
#define LDSS 40  // LDS row stride in halves (padded)

template <int MODE, bool HAS_BIAS>
__global__ __launch_bounds__(256)
void wmma_gemm_kernel(const f16* __restrict__ A, const f16* __restrict__ B,
                      const float* __restrict__ bias,
                      f16* __restrict__ out16, float* __restrict__ out32,
                      const float* __restrict__ res32,
                      const float* __restrict__ gamma,
                      int M, int N, int K) {
    __shared__ _Float16 As[64 * LDSS];
    __shared__ _Float16 Bt[64 * LDSS];  // transposed: Bt[n][k]

    const int tid  = threadIdx.x;
    const int wid  = tid >> 5;
    const int lane = tid & 31;
    const int nl   = lane & 15;
    const int hi   = lane >> 4;

    const int m0 = blockIdx.y * 64;
    const int n0 = blockIdx.x * 64;

    const int ctile = wid & 3;   // column 16-tile within block tile
    const int rgrp  = wid >> 2;  // row-tile group {0,1} -> rows rgrp*2 + {0,1}

    v8f acc0 = {0.f,0.f,0.f,0.f,0.f,0.f,0.f,0.f};
    v8f acc1 = {0.f,0.f,0.f,0.f,0.f,0.f,0.f,0.f};

    const int a_row  = tid >> 2;        // 0..63
    const int a_kc   = (tid & 3) << 3;  // 0,8,16,24
    const int b_krow = tid >> 3;        // 0..31
    const int b_nc   = (tid & 7) << 3;  // 0..56

    const f16* aptr = A + (size_t)(m0 + a_row) * K + a_kc;
    const f16* bptr = B + (size_t)b_krow * N + n0 + b_nc;

    for (int k0 = 0; k0 < K; k0 += 32) {
        h8 av = *(const h8*)(aptr + k0);
        h8 bv = *(const h8*)(bptr + (size_t)k0 * N);
        if (k0 + 32 < K) {   // prefetch next K-tile (global_prefetch_b8)
            __builtin_prefetch(aptr + k0 + 32, 0, 3);
            __builtin_prefetch(bptr + (size_t)(k0 + 32) * N, 0, 3);
        }
        __syncthreads();
        *(h8*)&As[a_row * LDSS + a_kc] = av;
#pragma unroll
        for (int i = 0; i < 8; ++i)
            Bt[(b_nc + i) * LDSS + b_krow] = bv[i];
        __syncthreads();

        // B fragment (32x16): lane -> col n, K striped 2/VGPR, +8 for hi lanes, +16 for VGPRs 4..7
        v16h bfrag;
#pragma unroll
        for (int v = 0; v < 8; ++v) {
            unsigned u = *(const unsigned*)&Bt[(ctile * 16 + nl) * LDSS +
                                               ((v >> 2) << 4) + (hi << 3) + ((v & 3) << 1)];
            frag_put2(bfrag, v, u);
        }
#pragma unroll
        for (int rt = 0; rt < 2; ++rt) {
            int row = (rgrp * 2 + rt) * 16 + nl;
            v16h afrag;
#pragma unroll
            for (int v = 0; v < 8; ++v) {
                int kb = ((v >> 2) << 4) + (hi << 3) + ((v & 3) << 1);
                unsigned u = *(const unsigned*)&As[row * LDSS + kb];
                frag_put2(afrag, v, u);
            }
            if (rt == 0)
                acc0 = __builtin_amdgcn_wmma_f32_16x16x32_f16(
                    false, afrag, false, bfrag, (short)0, acc0, false, false);
            else
                acc1 = __builtin_amdgcn_wmma_f32_16x16x32_f16(
                    false, afrag, false, bfrag, (short)0, acc1, false, false);
        }
    }

    // fused epilogue (straight-line; MODE/HAS_BIAS are compile-time)
    const int gn = n0 + ctile * 16 + nl;
    const float bcol = HAS_BIAS ? bias[gn] : 0.f;
    const float gcol = (MODE == 3) ? gamma[gn] : 0.f;
#pragma unroll
    for (int rt = 0; rt < 2; ++rt) {
#pragma unroll
        for (int r = 0; r < 8; ++r) {
            int gm = m0 + (rgrp * 2 + rt) * 16 + hi * 8 + r;
            float v = ((rt == 0) ? acc0[r] : acc1[r]) + bcol;
            size_t idx = (size_t)gm * N + gn;
            if (MODE == 0) {
                out16[idx] = (_Float16)v;
            } else if (MODE == 1) {
                float t = 0.7978845608028654f * (v + 0.044715f * v * v * v);
                out16[idx] = (_Float16)(0.5f * v * (1.0f + tanhf(t)));
            } else if (MODE == 2) {
                out32[idx] = res32[idx] + v;
            } else {
                out32[idx] = res32[idx] + gcol * v;
            }
        }
    }
}

// ---------------------------------------------------------------------------
// Windowed gather attention. One block (192 thr = 6 waves = 6 heads) per
// window w in [0, 4096). q rows padded 4->16 in the WMMA A operand.
// kv gather goes through GLOBAL_LOAD_ASYNC_TO_LDS_B128 (ASYNCcnt tracked).
// All pad guards are value-selects on clamped addresses: loads stay
// unconditional so EXEC never splits inside the WMMA pipeline.
// ---------------------------------------------------------------------------
__global__ __launch_bounds__(192)
void attn_kernel(const f16* __restrict__ q, const f16* __restrict__ kv,
                 const int* __restrict__ nh_idx, const int* __restrict__ mask,
                 f16* __restrict__ o) {
    __shared__ _Float16 ks[NHG * 384];
    __shared__ _Float16 vs[NHG * 384];
    __shared__ int   tok[NHG];
    __shared__ float maskf[NHG];
    __shared__ float scf[6 * 4 * NHG];       // raw scores (masked+scaled)
    __shared__ _Float16 attnH[6 * 4 * 64];   // probs, zero-padded to K=64

    const int w     = blockIdx.x;   // window index
    const int slice = w >> 10;      // (b,v,t) slice, NG = 1024
    const int g     = w & 1023;
    const int tid   = threadIdx.x;
    const int h     = tid >> 5;     // head = wave id
    const int lane  = tid & 31;
    const int nl    = lane & 15;
    const int hi    = lane >> 4;
    const bool qpad = (nl >= 4);    // padded query rows 4..15

    if (tid < NHG) {
        int t = slice * 4096 + nh_idx[g * NHG + tid];
        tok[tid]   = t;
        maskf[tid] = (mask[t] != 0) ? 1.f : 0.f;
    }
    __syncthreads();

    // async gather of k,v neighborhood rows straight into LDS (no VGPR bounce)
    for (int e = tid * 8; e < NHG * 384; e += 192 * 8) {
        int j = e / 384, c = e % 384;
        const f16* src = kv + (size_t)tok[j] * 768 + c;
        async_ld16(src,       &ks[e]);
        async_ld16(src + 384, &vs[e]);
    }
    async_wait0();       // this wave's ASYNCcnt -> 0
    __syncthreads();     // then all waves' gathers are visible

    // ---- scores = q_h(4x64) @ k_h^T(64x36), padded to 16x48, K-steps of 32
    const int qrow = qpad ? 3 : nl;                       // clamped address
    const f16* qrp = q + (size_t)(w * 4 + qrow) * 384 + h * 64;

    v8f s0 = {0.f,0.f,0.f,0.f,0.f,0.f,0.f,0.f};
    v8f s1 = s0, s2 = s0;
    for (int kstep = 0; kstep < 2; ++kstep) {
        v16h a;
#pragma unroll
        for (int v = 0; v < 8; ++v) {
            int kb = kstep * 32 + ((v >> 2) << 4) + (hi << 3) + ((v & 3) << 1);
            unsigned u = *(const unsigned*)(qrp + kb);     // unconditional
            if (qpad) u = 0;                               // v_cndmask
            frag_put2(a, v, u);
        }
#pragma unroll
        for (int t = 0; t < 3; ++t) {
            int mkey = t * 16 + nl;
            int mclp = (mkey < NHG) ? mkey : (NHG - 1);    // clamped address
            const _Float16* krp = &ks[mclp * 384 + h * 64];
            v16h b;
#pragma unroll
            for (int v = 0; v < 8; ++v) {
                int kd = kstep * 32 + ((v >> 2) << 4) + (hi << 3) + ((v & 3) << 1);
                unsigned u = *(const unsigned*)(krp + kd); // unconditional
                if (mkey >= NHG) u = 0;                    // v_cndmask
                frag_put2(b, v, u);
            }
            if (t == 0)
                s0 = __builtin_amdgcn_wmma_f32_16x16x32_f16(false, a, false, b, (short)0, s0, false, false);
            else if (t == 1)
                s1 = __builtin_amdgcn_wmma_f32_16x16x32_f16(false, a, false, b, (short)0, s1, false, false);
            else
                s2 = __builtin_amdgcn_wmma_f32_16x16x32_f16(false, a, false, b, (short)0, s2, false, false);
        }
    }

    // scale + mask, rows 0..3 live in lanes 0..15 (hi==0), element r
    if (hi == 0) {
#pragma unroll
        for (int r = 0; r < 4; ++r) {
#pragma unroll
            for (int t = 0; t < 3; ++t) {
                int col = t * 16 + nl;
                if (col < NHG) {
                    float sv = (t == 0 ? s0[r] : (t == 1 ? s1[r] : s2[r])) * 0.125f;
                    if (maskf[col] != 0.f) sv = -1e9f;
                    scf[(h * 4 + r) * NHG + col] = sv;
                }
            }
        }
    }
    __syncthreads();

    // softmax over 36 keys; lanes 0..3 of each wave handle the 4 query rows
    if (lane < 4) {
        float* sr = &scf[(h * 4 + lane) * NHG];
        float mx = -3.4e38f;
        for (int c = 0; c < NHG; ++c) mx = fmaxf(mx, sr[c]);
        float den = 0.f;
        for (int c = 0; c < NHG; ++c) { float e = __expf(sr[c] - mx); sr[c] = e; den += e; }
        float inv = 1.f / den;
        _Float16* ar = &attnH[(h * 4 + lane) * 64];
        for (int c = 0; c < 64; ++c)
            ar[c] = (c < NHG) ? (_Float16)(sr[c] * inv) : (_Float16)0.f;
    }
    __syncthreads();

    // ---- o = attn(4x36) @ v(36x64), attn zero-padded to 16x64
    const _Float16* arp = &attnH[(h * 4 + qrow) * 64];     // clamped address

    v8f o0 = {0.f,0.f,0.f,0.f,0.f,0.f,0.f,0.f};
    v8f o1 = o0, o2 = o0, o3 = o0;
    for (int kstep = 0; kstep < 2; ++kstep) {
        v16h a;
#pragma unroll
        for (int v = 0; v < 8; ++v) {
            int kb = kstep * 32 + ((v >> 2) << 4) + (hi << 3) + ((v & 3) << 1);
            unsigned u = *(const unsigned*)(arp + kb);     // unconditional
            if (qpad) u = 0;                               // v_cndmask
            frag_put2(a, v, u);
        }
#pragma unroll
        for (int t = 0; t < 4; ++t) {
            int d = t * 16 + nl;
            v16h b;
#pragma unroll
            for (int v = 0; v < 8; ++v) {
                int kk  = kstep * 32 + ((v >> 2) << 4) + (hi << 3) + ((v & 3) << 1);
                int k0c = (kk     < NHG) ? kk     : 0;     // clamped addresses
                int k1c = (kk + 1 < NHG) ? kk + 1 : 0;
                _Float16 b0 = vs[k0c * 384 + h * 64 + d];  // unconditional
                _Float16 b1 = vs[k1c * 384 + h * 64 + d];
                if (kk     >= NHG) b0 = (_Float16)0.f;     // v_cndmask
                if (kk + 1 >= NHG) b1 = (_Float16)0.f;
                b[2 * v]     = b0;
                b[2 * v + 1] = b1;
            }
            if (t == 0)
                o0 = __builtin_amdgcn_wmma_f32_16x16x32_f16(false, a, false, b, (short)0, o0, false, false);
            else if (t == 1)
                o1 = __builtin_amdgcn_wmma_f32_16x16x32_f16(false, a, false, b, (short)0, o1, false, false);
            else if (t == 2)
                o2 = __builtin_amdgcn_wmma_f32_16x16x32_f16(false, a, false, b, (short)0, o2, false, false);
            else
                o3 = __builtin_amdgcn_wmma_f32_16x16x32_f16(false, a, false, b, (short)0, o3, false, false);
        }
    }

    if (hi == 0) {
#pragma unroll
        for (int r = 0; r < 4; ++r) {
#pragma unroll
            for (int t = 0; t < 4; ++t) {
                float vo = (t == 0 ? o0[r] : (t == 1 ? o1[r] : (t == 2 ? o2[r] : o3[r])));
                o[(size_t)(w * 4 + r) * 384 + h * 64 + t * 16 + nl] = (_Float16)vo;
            }
        }
    }
}

// ---------------------------------------------------------------------------
extern "C" void kernel_launch(void* const* d_in, const int* in_sizes, int n_in,
                              void* d_out, int out_size, void* d_ws, size_t ws_size,
                              hipStream_t stream) {
    (void)in_sizes; (void)n_in; (void)out_size; (void)ws_size;

    const float* x       = (const float*)d_in[0];
    const int*   mask    = (const int*)d_in[1];
    const int*   nh_idx  = (const int*)d_in[2];
    const float* ln_q_s  = (const float*)d_in[3];
    const float* ln_q_b  = (const float*)d_in[4];
    const float* Wq      = (const float*)d_in[5];
    const float* ln_kv_s = (const float*)d_in[6];
    const float* ln_kv_b = (const float*)d_in[7];
    const float* Wkv     = (const float*)d_in[8];
    const float* bkv     = (const float*)d_in[9];
    const float* Wo      = (const float*)d_in[10];
    const float* ln_m_s  = (const float*)d_in[11];
    const float* ln_m_b  = (const float*)d_in[12];
    const float* W_emb   = (const float*)d_in[13];
    const float* b_emb   = (const float*)d_in[14];
    const float* W1      = (const float*)d_in[15];
    const float* b1      = (const float*)d_in[16];
    const float* W2      = (const float*)d_in[17];
    const float* b2      = (const float*)d_in[18];
    const float* gamma   = (const float*)d_in[19];

    const size_t M = 16384, C = 384, C2 = 768;
    char* base = (char*)d_ws;
    size_t off = 0;
    auto alloc = [&](size_t bytes) {
        size_t o = off; off += (bytes + 255) & ~(size_t)255; return o;
    };

    f16* wq16   = (f16*)(base + alloc(C * C  * 2));
    f16* wkv16  = (f16*)(base + alloc(C * C2 * 2));
    f16* wo16   = (f16*)(base + alloc(C * C  * 2));
    f16* wemb16 = (f16*)(base + alloc(C * C  * 2));
    f16* w116   = (f16*)(base + alloc(C * C2 * 2));
    f16* w216   = (f16*)(base + alloc(C2 * C * 2));
    f16* qin    = (f16*)(base + alloc(M * C  * 2));  // reused as xm
    f16* kvin   = (f16*)(base + alloc(M * C  * 2));  // reused as xemb
    f16* qbuf   = (f16*)(base + alloc(M * C  * 2));
    f16* kvbuf  = (f16*)(base + alloc(M * C2 * 2));  // reused as h
    f16* obuf   = (f16*)(base + alloc(M * C  * 2));
    float* x1   = (float*)(base + alloc(M * C * 4));

    auto cvt = [&](const float* s, f16* d, int n) {
        f32_to_f16_kernel<<<dim3((n + 255) / 256), dim3(256), 0, stream>>>(s, d, n);
    };
    cvt(Wq, wq16, (int)(C * C));
    cvt(Wkv, wkv16, (int)(C * C2));
    cvt(Wo, wo16, (int)(C * C));
    cvt(W_emb, wemb16, (int)(C * C));
    cvt(W1, w116, (int)(C * C2));
    cvt(W2, w216, (int)(C2 * C));

    // LN (q-path and kv-path in one pass)
    ln_dual_kernel<true><<<dim3(M / 8), dim3(256), 0, stream>>>(
        x, ln_q_s, ln_q_b, ln_kv_s, ln_kv_b, qin, kvin, (int)M);

    // q = LNq(x) @ Wq
    wmma_gemm_kernel<0, false><<<dim3(C / 64, M / 64), dim3(256), 0, stream>>>(
        qin, wq16, nullptr, qbuf, nullptr, nullptr, nullptr,
        (int)M, (int)C, (int)C);
    // kv = LNkv(x) @ Wkv + bkv
    wmma_gemm_kernel<0, true><<<dim3(C2 / 64, M / 64), dim3(256), 0, stream>>>(
        kvin, wkv16, bkv, kvbuf, nullptr, nullptr, nullptr,
        (int)M, (int)C2, (int)C);

    // windowed gather attention -> o (f16)
    attn_kernel<<<dim3(4096), dim3(192), 0, stream>>>(qbuf, kvbuf, nh_idx, mask, obuf);

    // x1 = x + o @ Wo
    wmma_gemm_kernel<2, false><<<dim3(C / 64, M / 64), dim3(256), 0, stream>>>(
        obuf, wo16, nullptr, nullptr, x1, x, nullptr,
        (int)M, (int)C, (int)C);

    // MLP block
    f16* xm = qin; f16* xemb = kvin; f16* hbuf = kvbuf;
    ln_dual_kernel<false><<<dim3(M / 8), dim3(256), 0, stream>>>(
        x1, ln_m_s, ln_m_b, nullptr, nullptr, xm, nullptr, (int)M);
    wmma_gemm_kernel<0, true><<<dim3(C / 64, M / 64), dim3(256), 0, stream>>>(
        xm, wemb16, b_emb, xemb, nullptr, nullptr, nullptr,
        (int)M, (int)C, (int)C);
    wmma_gemm_kernel<1, true><<<dim3(C2 / 64, M / 64), dim3(256), 0, stream>>>(
        xemb, w116, b1, hbuf, nullptr, nullptr, nullptr,
        (int)M, (int)C2, (int)C);
    // out = x1 + gamma * (h @ W2 + b2)
    wmma_gemm_kernel<3, true><<<dim3(C / 64, M / 64), dim3(256), 0, stream>>>(
        hbuf, w216, b2, nullptr, (float*)d_out, x1, gamma,
        (int)M, (int)C, (int)C2);
}
